// GaussianKernel3_84524956385428
// MI455X (gfx1250) — matrix operand order repaired and checked by reading, not compiled, not run on hardware
//
#include <hip/hip_runtime.h>

typedef __attribute__((ext_vector_type(16))) __bf16 v16bf;
typedef __attribute__((ext_vector_type(8)))  float  v8f;

// ---------------------------------------------------------------------------
// fp32 -> bf16 (round-to-nearest-even) helpers
// ---------------------------------------------------------------------------
__device__ __forceinline__ unsigned int bf16rne(float f) {
  unsigned int u = __float_as_uint(f);
  return (u + 0x7FFFu + ((u >> 16) & 1u)) >> 16;   // low 16 bits hold bf16
}
__device__ __forceinline__ float bf16hi_to_f32(unsigned int b) {
  return __uint_as_float(b << 16);
}

union FragAB {                 // 8 dwords = 16 bf16 = one WMMA A/B operand
  uint4 q[2];
  v16bf v;
};

// ---------------------------------------------------------------------------
// Pre-pass 1: exact fp32 row squared-norms, one wave32 per row (F = 512)
// ---------------------------------------------------------------------------
__global__ __launch_bounds__(256) void row_sqnorm_kernel(
    const float* __restrict__ src, float* __restrict__ dst, int rows) {
  int wave = (int)(blockIdx.x * (blockDim.x >> 5)) + (threadIdx.x >> 5);
  int lane = threadIdx.x & 31;
  if (wave >= rows) return;
  const float4* r = (const float4*)(src + (size_t)wave * 512);  // 128 float4
  float s = 0.f;
#pragma unroll
  for (int i = 0; i < 4; ++i) {
    float4 v = r[lane + 32 * i];
    s += v.x * v.x + v.y * v.y + v.z * v.z + v.w * v.w;
  }
#pragma unroll
  for (int off = 16; off; off >>= 1) s += __shfl_xor(s, off, 32);
  if (lane == 0) dst[wave] = s;
}

// ---------------------------------------------------------------------------
// Pre-pass 2: split fp32 -> bf16 hi + bf16 lo (residual), 8 elements/thread.
// dh/dl are uint4 arrays: one uint4 = 8 packed bf16.
// ---------------------------------------------------------------------------
__global__ __launch_bounds__(256) void split_bf16_kernel(
    const float* __restrict__ src, uint4* __restrict__ dh,
    uint4* __restrict__ dl, int n8) {
  int i = (int)(blockIdx.x * blockDim.x) + threadIdx.x;
  if (i >= n8) return;
  const float4* s = (const float4*)src + (size_t)i * 2;
  float4 a = s[0], b = s[1];
  unsigned int h0 = bf16rne(a.x), h1 = bf16rne(a.y), h2 = bf16rne(a.z), h3 = bf16rne(a.w);
  unsigned int h4 = bf16rne(b.x), h5 = bf16rne(b.y), h6 = bf16rne(b.z), h7 = bf16rne(b.w);
  float r0 = a.x - bf16hi_to_f32(h0), r1 = a.y - bf16hi_to_f32(h1);
  float r2 = a.z - bf16hi_to_f32(h2), r3 = a.w - bf16hi_to_f32(h3);
  float r4 = b.x - bf16hi_to_f32(h4), r5 = b.y - bf16hi_to_f32(h5);
  float r6 = b.z - bf16hi_to_f32(h6), r7 = b.w - bf16hi_to_f32(h7);
  dh[i] = make_uint4(h0 | (h1 << 16), h2 | (h3 << 16),
                     h4 | (h5 << 16), h6 | (h7 << 16));
  dl[i] = make_uint4(bf16rne(r0) | (bf16rne(r1) << 16),
                     bf16rne(r2) | (bf16rne(r3) << 16),
                     bf16rne(r4) | (bf16rne(r5) << 16),
                     bf16rne(r6) | (bf16rne(r7) << 16));
}

// ---------------------------------------------------------------------------
// Fused kernel: bf16x3 split GEMM (x · w^T) on v_wmma_f32_16x16x32_bf16,
// epilogue computes exp(-gamma * (x2 + w2 - 2*xw)).
//
// Tiling: 256 threads (8 wave32), 128x128 C tile, BK = 32.
// Wave grid 2x4: each wave owns 64 rows x 32 cols = 4x2 WMMA tiles.
// PRECONV=true : tiles streamed from pre-split bf16 arrays (no VALU convert).
// PRECONV=false: tiles converted fp32->bf16 hi/lo in the loader (fallback).
// ---------------------------------------------------------------------------
#define LDS_STRIDE 20   // dwords/row: 16 data (32 bf16) + 4 pad (80B, 16B-aligned)

template <bool PRECONV>
__global__ __launch_bounds__(256) void gauss_wmma_kernel(
    const float* __restrict__ x,      // [N,512] fp32
    const float* __restrict__ w,      // [L,512] fp32
    const float* __restrict__ gamma_elm,
    const float* __restrict__ x2,     // [N]
    const float* __restrict__ w2,     // [L]
    const uint4* __restrict__ xh, const uint4* __restrict__ xl,  // bf16 splits
    const uint4* __restrict__ wh, const uint4* __restrict__ wl,  // (PRECONV)
    float* __restrict__ out,          // [N,L]
    int L) {
  __shared__ unsigned int sAh[128 * LDS_STRIDE];
  __shared__ unsigned int sAl[128 * LDS_STRIDE];
  __shared__ unsigned int sBh[128 * LDS_STRIDE];
  __shared__ unsigned int sBl[128 * LDS_STRIDE];

  const int tid    = threadIdx.x;
  const int lane   = tid & 31;
  const int waveId = tid >> 5;
  const int wm = waveId >> 2;          // 0..1 : 64-row strip
  const int wn = waveId & 3;           // 0..3 : 32-col strip
  const int lrow = lane & 15;
  const int lhi  = lane >> 4;          // upper/lower half-wave

  const int rowBlk = blockIdx.y * 128; // x rows
  const int colBlk = blockIdx.x * 128; // landmarks

  v8f acc[4][2];
#pragma unroll
  for (int tm = 0; tm < 4; ++tm)
#pragma unroll
    for (int tn = 0; tn < 2; ++tn)
      acc[tm][tn] = v8f{0.f, 0.f, 0.f, 0.f, 0.f, 0.f, 0.f, 0.f};

  const int kdwA = lhi * 4;            // A frag: run1 dword offset (K 0..7 / 8..15)

  for (int kc = 0; kc < 16; ++kc) {    // 512 / 32 K-chunks
    __syncthreads();                   // LDS safe to overwrite

    if constexpr (PRECONV) {
      // ---- stream pre-split bf16 tiles -> LDS (row = 64 uint4 in global)
#pragma unroll
      for (int half = 0; half < 2; ++half) {
        const uint4* sh = (half == 0 ? xh + (size_t)rowBlk * 64
                                     : wh + (size_t)colBlk * 64);
        const uint4* sl = (half == 0 ? xl + (size_t)rowBlk * 64
                                     : wl + (size_t)colBlk * 64);
        unsigned int* dh = (half == 0 ? sAh : sBh);
        unsigned int* dl = (half == 0 ? sAl : sBl);
#pragma unroll
        for (int i = 0; i < 2; ++i) {
          int idx = tid + i * 256;             // 0..511
          int row = idx >> 2;                  // 0..127
          int q   = idx & 3;                   // 16B chunk within 64B row-slice
          const uint4* ph = sh + (size_t)row * 64 + kc * 4 + q;
          const uint4* pl = sl + (size_t)row * 64 + kc * 4 + q;
          uint4 vh = *ph;
          uint4 vl = *pl;
          __builtin_prefetch(ph + 4, 0, 3);    // next K-chunk (global_prefetch_b8)
          *(uint4*)&dh[row * LDS_STRIDE + q * 4] = vh;
          *(uint4*)&dl[row * LDS_STRIDE + q * 4] = vl;
        }
      }
    } else {
      // ---- fallback: fp32 tiles -> bf16 hi/lo in-register split -> LDS
#pragma unroll
      for (int half = 0; half < 2; ++half) {
        const float* src = (half == 0 ? x + (size_t)rowBlk * 512
                                      : w + (size_t)colBlk * 512) + kc * 32;
        unsigned int* dh = (half == 0 ? sAh : sBh);
        unsigned int* dl = (half == 0 ? sAl : sBl);
#pragma unroll
        for (int i = 0; i < 4; ++i) {
          int idx = tid + i * 256;             // 0..1023
          int row = idx >> 3;                  // 0..127
          int kq  = idx & 7;                   // float4 within 32-K chunk
          const float* p = src + (size_t)row * 512 + kq * 4;
          float4 v = *(const float4*)p;
          __builtin_prefetch(p + 32, 0, 3);    // next K-chunk
          unsigned int h0 = bf16rne(v.x), h1 = bf16rne(v.y);
          unsigned int h2 = bf16rne(v.z), h3 = bf16rne(v.w);
          float r0 = v.x - bf16hi_to_f32(h0), r1 = v.y - bf16hi_to_f32(h1);
          float r2 = v.z - bf16hi_to_f32(h2), r3 = v.w - bf16hi_to_f32(h3);
          *(uint2*)&dh[row * LDS_STRIDE + kq * 2] =
              make_uint2(h0 | (h1 << 16), h2 | (h3 << 16));
          *(uint2*)&dl[row * LDS_STRIDE + kq * 2] =
              make_uint2(bf16rne(r0) | (bf16rne(r1) << 16),
                         bf16rne(r2) | (bf16rne(r3) << 16));
        }
      }
    }
    __syncthreads();

    // ---- A fragments for this wave's 4 row tiles (ISA 16-bit 16x32 layout)
    FragAB ah[4], al[4];
#pragma unroll
    for (int tm = 0; tm < 4; ++tm) {
      const unsigned int* ph = &sAh[(wm * 64 + tm * 16 + lrow) * LDS_STRIDE];
      const unsigned int* pl = &sAl[(wm * 64 + tm * 16 + lrow) * LDS_STRIDE];
      ah[tm].q[0] = *(const uint4*)(ph + kdwA);       // K kb..kb+7
      ah[tm].q[1] = *(const uint4*)(ph + kdwA + 8);   // K kb+16..kb+23
      al[tm].q[0] = *(const uint4*)(pl + kdwA);
      al[tm].q[1] = *(const uint4*)(pl + kdwA + 8);
    }

    // ---- B fragments + bf16x3 WMMA accumulation
#pragma unroll
    for (int tn = 0; tn < 2; ++tn) {
      const unsigned int* ph =
          &sBh[(wn * 32 + tn * 16 + lrow) * LDS_STRIDE + lhi * 4];
      const unsigned int* pl =
          &sBl[(wn * 32 + tn * 16 + lrow) * LDS_STRIDE + lhi * 4];
      FragAB bh, bl;
      bh.q[0] = *(const uint4*)(ph);     // K 0..7,16..23 (lanes 0-15) / 8..15,24..31 (16-31)
      bh.q[1] = *(const uint4*)(ph + 8);
      bl.q[0] = *(const uint4*)(pl);
      bl.q[1] = *(const uint4*)(pl + 8);
#pragma unroll
      for (int tm = 0; tm < 4; ++tm) {
        acc[tm][tn] = __builtin_amdgcn_wmma_f32_16x16x32_bf16(
            false, ah[tm].v, false, bh.v, (short)0, acc[tm][tn], false, false);
        acc[tm][tn] = __builtin_amdgcn_wmma_f32_16x16x32_bf16(
            false, al[tm].v, false, bh.v, (short)0, acc[tm][tn], false, false);
        acc[tm][tn] = __builtin_amdgcn_wmma_f32_16x16x32_bf16(
            false, ah[tm].v, false, bl.v, (short)0, acc[tm][tn], false, false);
      }
    }
  }

  // ---- epilogue: d2 = x2 + w2 - 2*xw ; out = exp(-gamma*d2)
  const float mg = -expf(gamma_elm[0]);

#pragma unroll
  for (int tm = 0; tm < 4; ++tm) {
    const int rowBase = rowBlk + wm * 64 + tm * 16 + lhi * 8;  // +r, r=0..7
    float xs[8];
#pragma unroll
    for (int r = 0; r < 8; ++r) xs[r] = x2[rowBase + r];
#pragma unroll
    for (int tn = 0; tn < 2; ++tn) {
      const int col = colBlk + wn * 32 + tn * 16 + lrow;
      const float w2v = w2[col];
      v8f c = acc[tm][tn];
#pragma unroll
      for (int r = 0; r < 8; ++r) {
        float d2 = xs[r] + w2v - 2.0f * c[r];
        out[(size_t)(rowBase + r) * (size_t)L + col] = expf(mg * d2);
      }
    }
  }
}

// ---------------------------------------------------------------------------
// Launch: norms (+ optional bf16 split pre-pass) into workspace, then GEMM
// ---------------------------------------------------------------------------
extern "C" void kernel_launch(void* const* d_in, const int* in_sizes, int n_in,
                              void* d_out, int out_size, void* d_ws, size_t ws_size,
                              hipStream_t stream) {
  const float* x = (const float*)d_in[0];   // [N,512]
  const float* w = (const float*)d_in[1];   // [L,512]
  const float* g = (const float*)d_in[2];   // [1]
  float* out = (float*)d_out;

  const int N = in_sizes[0] / 512;          // 16384
  const int L = in_sizes[1] / 512;          // 2048

  // workspace layout (16B-aligned slices)
  char* p = (char*)d_ws;
  float* x2 = (float*)p;            p += (size_t)N * sizeof(float);
  float* w2 = (float*)p;            p += (size_t)L * sizeof(float);
  uint4* xh = (uint4*)p;            p += (size_t)N * 512 * 2;   // bf16
  uint4* xl = (uint4*)p;            p += (size_t)N * 512 * 2;
  uint4* wh = (uint4*)p;            p += (size_t)L * 512 * 2;
  uint4* wl = (uint4*)p;            p += (size_t)L * 512 * 2;
  const size_t need = (size_t)(p - (char*)d_ws);
  const bool preconv = (ws_size >= need);   // constant per process -> deterministic

  // exact fp32 row norms: one wave32 per row, 8 waves per block
  row_sqnorm_kernel<<<(N + 7) / 8, 256, 0, stream>>>(x, x2, N);
  row_sqnorm_kernel<<<(L + 7) / 8, 256, 0, stream>>>(w, w2, L);

  dim3 grid(L / 128, N / 128);              // (16, 128)
  if (preconv) {
    const int nx8 = N * 64;                 // N*512/8
    const int nw8 = L * 64;
    split_bf16_kernel<<<(nx8 + 255) / 256, 256, 0, stream>>>(x, xh, xl, nx8);
    split_bf16_kernel<<<(nw8 + 255) / 256, 256, 0, stream>>>(w, wh, wl, nw8);
    gauss_wmma_kernel<true><<<grid, dim3(256), 0, stream>>>(
        x, w, g, x2, w2, xh, xl, wh, wl, out, L);
  } else {
    gauss_wmma_kernel<false><<<grid, dim3(256), 0, stream>>>(
        x, w, g, x2, w2, nullptr, nullptr, nullptr, nullptr, out, L);
  }
}
